// SO2_Convolution_79439715106831
// MI455X (gfx1250) — matrix operand order repaired and checked by reading, not compiled, not run on hardware
//
#include <hip/hip_runtime.h>

#define L2S 25
#define CH  128
#define TM  32

typedef __attribute__((ext_vector_type(16))) __bf16 v16bf;
typedef __attribute__((ext_vector_type(8)))  float  v8f;

union BFrag {
  v16bf v;
  unsigned short u16[16];
  uint4 q[2];
};

__device__ __forceinline__ unsigned short f2bf(float f) {
  unsigned int u = __float_as_uint(f);
  u += 0x7FFFu + ((u >> 16) & 1u);   // round-to-nearest-even
  return (unsigned short)(u >> 16);
}

__device__ __forceinline__ v8f vzero() {
  v8f z = {0.f, 0.f, 0.f, 0.f, 0.f, 0.f, 0.f, 0.f};
  return z;
}

__device__ __forceinline__ v8f wmma_bf16(v16bf a, v16bf b, v8f c) {
  return __builtin_amdgcn_wmma_f32_16x16x32_bf16(false, a, false, b, (short)0, c,
                                                 false, false);
}

// A-matrix 16x32 bf16 fragment from LDS (row-major, `stride` elems per row).
// lane row = lane%16; lanes0-15: K = {kb..kb+7, kb+16..kb+23}; lanes16-31: +8.
__device__ __forceinline__ v16bf load_a_lds(const unsigned short* lds, int stride,
                                            int kbase, int lr, int lh) {
  BFrag f;
  const unsigned short* p = lds + lr * stride + kbase + 8 * lh;
  f.q[0] = *(const uint4*)(p);
  f.q[1] = *(const uint4*)(p + 16);
  return f.v;
}

// B fragment from pre-packed bf16 weights: fragment (tile tn, kstep k) is
// stored as 32 lanes x 16 contiguous bf16 -> two coalesced b128 loads, no VALU.
__device__ __forceinline__ v16bf load_b_packed(const unsigned short* __restrict__ P,
                                               int KS, int tn, int k, int lane) {
  BFrag f;
  const unsigned short* p = P + ((size_t)((tn * KS + k) * 32 + lane) << 4);
  f.q[0] = *(const uint4*)(p);
  f.q[1] = *(const uint4*)(p + 8);
  return f.v;
}

// Pack a row-major fp32 weight [K][N] into WMMA B-fragment order, bf16.
// One thread per (tile, kstep, lane) -> 16 elements.
__global__ __launch_bounds__(256) void pack_weights_kernel(
    const float* __restrict__ W, unsigned short* __restrict__ P, int K, int N) {
  int idx = blockIdx.x * 256 + threadIdx.x;
  int total = (N / 16) * (K / 32) * 32;
  if (idx >= total) return;
  int lane = idx & 31;
  int rem = idx >> 5;
  int KS = K / 32;
  int k = rem % KS;
  int t = rem / KS;
  int lr = lane & 15, lh = lane >> 4;
  const float* src = W + (size_t)(k * 32 + 8 * lh) * N + t * 16 + lr;
  unsigned short* dst = P + ((size_t)idx << 4);
#pragma unroll
  for (int i = 0; i < 16; i++) dst[i] = f2bf(src[(size_t)(i < 8 ? i : i + 8) * N]);
}

// Gate epilogue: add bias, gate the gathered embedding rows, store bf16 A rows.
__device__ __forceinline__ void gate_epilogue(
    v8f g, int t, int rt, int mi, const float* __restrict__ bd,
    const float* __restrict__ embBase, unsigned short* lds_a,
    int S, int setStride, int nsets, int lr, int lh) {
  int c = t * 16 + lr;
  float bias = bd[c];
  int j = c >> 7, ch = c & 127;
  int l = j + mi;
  int slotP = l * l + l + mi;
  int slotM = l * l + l - mi;
  for (int s = 0; s < nsets; s++) {
    const float* ep = embBase + (size_t)((s == 0) ? slotP : slotM) * CH + ch;
    unsigned short* ap = lds_a + s * setStride + c;
#pragma unroll
    for (int i = 0; i < 8; i++) {
      int row = rt * 16 + i + 8 * lh;
      ap[row * S] = f2bf(ep[(size_t)row * L2S * CH] * (g[i] + bias));
    }
  }
}

__global__ __launch_bounds__(256) void so2_conv_kernel(
    const float* __restrict__ emb, const float* __restrict__ xe,
    const unsigned short* __restrict__ pWd0, const float* __restrict__ bd0,
    const unsigned short* __restrict__ pW0,  const float* __restrict__ b0,
    const unsigned short* __restrict__ pWd1, const float* __restrict__ bd1,
    const unsigned short* __restrict__ pW1,
    const unsigned short* __restrict__ pWd2, const float* __restrict__ bd2,
    const unsigned short* __restrict__ pW2,
    float* __restrict__ out)
{
  __shared__ __align__(16) unsigned short lds_xe[TM * 136];        // x_edge bf16
  __shared__ __align__(16) unsigned short lds_a[2 * TM * 520];     // gated A, bf16

  const int tid  = threadIdx.x;
  const int wave = tid >> 5;
  const int lane = tid & 31;
  const int lr   = lane & 15;
  const int lh   = lane >> 4;
  const int node0 = blockIdx.x * TM;

  float* outBase = out + (size_t)node0 * L2S * CH;
  const float* embBase = emb + (size_t)node0 * L2S * CH;

  // ---- zero the |m|>2 output slots: 9,15,16,17,23,24 (out is poisoned) ----
  for (int idx = tid; idx < TM * CH; idx += 256) {
    int row = idx >> 7, ch = idx & 127;
    size_t ro = (size_t)row * L2S * CH + ch;
    outBase[ro +  9 * CH] = 0.f;
    outBase[ro + 15 * CH] = 0.f;
    outBase[ro + 16 * CH] = 0.f;
    outBase[ro + 17 * CH] = 0.f;
    outBase[ro + 23 * CH] = 0.f;
    outBase[ro + 24 * CH] = 0.f;
  }

  // ---- stage x_edge tile to LDS as bf16 (padded stride 136) ----
  for (int idx = tid; idx < TM * CH; idx += 256) {
    int row = idx >> 7, k = idx & 127;
    lds_xe[row * 136 + k] = f2bf(xe[(size_t)(node0 + row) * CH + k]);
  }
  __syncthreads();

  const unsigned short* pWdArr[3] = {pWd0, pWd1, pWd2};
  const float*          bdArr[3]  = {bd0, bd1, bd2};
  const unsigned short* pWArr[3]  = {pW0, pW1, pW2};

#pragma unroll
  for (int mi = 0; mi < 3; mi++) {
    const int ml = 5 - mi;           // 5, 4, 3
    const int K  = ml * CH;          // 640, 512, 384
    const int S  = K + 8;            // padded LDS row stride (16B-aligned rows)
    const int setStride = TM * S;
    const int nsets = (mi == 0) ? 1 : 2;
    const unsigned short* pWd = pWdArr[mi];
    const float* bd = bdArr[mi];
    const unsigned short* pW = pWArr[mi];

    if (mi) __syncthreads();         // lds_a reuse across m paths

    // ---- gate GEMM: g = xe @ Wd + bd, then A = bf16(emb * g) into LDS ----
    // Two column tiles per iteration (shared A fragment, 2 WMMA chains).
    const int gh = K / 16;                         // 40, 32, 24 (units per half)
    for (int u = wave; u < gh; u += 8) {
      int t = u >> 1, rt = u & 1;
      int t2 = t + K / 32;                         // second column tile
      const unsigned short* xebase = lds_xe + rt * 16 * 136;
      v8f gA = vzero(), gB = vzero();
#pragma unroll
      for (int k = 0; k < 4; k++) {
        v16bf a  = load_a_lds(xebase, 136, 32 * k, lr, lh);
        v16bf bA = load_b_packed(pWd, 4, t,  k, lane);
        v16bf bB = load_b_packed(pWd, 4, t2, k, lane);
        gA = wmma_bf16(a, bA, gA);
        gB = wmma_bf16(a, bB, gB);
      }
      gate_epilogue(gA, t,  rt, mi, bd, embBase, lds_a, S, setStride, nsets, lr, lh);
      gate_epilogue(gB, t2, rt, mi, bd, embBase, lds_a, S, setStride, nsets, lr, lh);
    }
    __syncthreads();

    // ---- main GEMM + recombination ----
    const int ksteps = K / 32;       // 20, 16, 12
    if (mi == 0) {
      // Two column tiles x two row tiles per iteration -> 4 WMMA chains.
      for (int t2i = wave; t2i < 20; t2i += 8) {
        int tt[2] = {t2i, t2i + 20};
        v8f acc[2][2];
#pragma unroll
        for (int p = 0; p < 2; p++) { acc[p][0] = vzero(); acc[p][1] = vzero(); }
        for (int k = 0; k < ksteps; k++) {
          v16bf ba = load_b_packed(pW, 20, tt[0], k, lane);
          v16bf bb = load_b_packed(pW, 20, tt[1], k, lane);
#pragma unroll
          for (int rt = 0; rt < 2; rt++) {
            v16bf a = load_a_lds(lds_a + rt * 16 * S, S, 32 * k, lr, lh);
            acc[0][rt] = wmma_bf16(a, ba, acc[0][rt]);
            acc[1][rt] = wmma_bf16(a, bb, acc[1][rt]);
          }
        }
#pragma unroll
        for (int p = 0; p < 2; p++) {
          int c = tt[p] * 16 + lr;
          int j = c >> 7, ch = c & 127;
          float bias = b0[c];
          float* op = outBase + (size_t)(j * (j + 1)) * CH + ch;   // slot l^2+l
#pragma unroll
          for (int rt = 0; rt < 2; rt++)
#pragma unroll
            for (int i = 0; i < 8; i++)
              op[(size_t)(rt * 16 + i + 8 * lh) * L2S * CH] = acc[p][rt][i] + bias;
        }
      }
    } else {
      // paired column tiles: (slot j, channel base k0) covers h=0 and h=1
      // 8 independent WMMA chains per iteration -> no D->C hazard stalls.
      for (int t = wave; t < ml * 8; t += 8) {
        int c0 = t * 16;
        int j = c0 >> 7, k0 = c0 & 127;
        int tn0 = j * 16 + (k0 >> 4);              // h=0 tile; h=1 is +8
        v8f yp0[2], ym0[2], yp1[2], ym1[2];
#pragma unroll
        for (int rt = 0; rt < 2; rt++) {
          yp0[rt] = vzero(); ym0[rt] = vzero();
          yp1[rt] = vzero(); ym1[rt] = vzero();
        }
        for (int k = 0; k < ksteps; k++) {
          v16bf bb0 = load_b_packed(pW, ksteps, tn0,     k, lane);
          v16bf bb1 = load_b_packed(pW, ksteps, tn0 + 8, k, lane);
#pragma unroll
          for (int rt = 0; rt < 2; rt++) {
            v16bf ap = load_a_lds(lds_a + rt * 16 * S,             S, 32 * k, lr, lh);
            v16bf am = load_a_lds(lds_a + setStride + rt * 16 * S, S, 32 * k, lr, lh);
            yp0[rt] = wmma_bf16(ap, bb0, yp0[rt]);
            ym0[rt] = wmma_bf16(am, bb0, ym0[rt]);
            yp1[rt] = wmma_bf16(ap, bb1, yp1[rt]);
            ym1[rt] = wmma_bf16(am, bb1, ym1[rt]);
          }
        }
        int l = j + mi;
        int slotP = l * l + l + mi;
        int slotM = l * l + l - mi;
        int ch = k0 + lr;
        float* opr = outBase + (size_t)slotP * CH + ch;
        float* opi = outBase + (size_t)slotM * CH + ch;
#pragma unroll
        for (int rt = 0; rt < 2; rt++)
#pragma unroll
          for (int i = 0; i < 8; i++) {
            size_t off = (size_t)(rt * 16 + i + 8 * lh) * L2S * CH;
            opr[off] = yp0[rt][i] - ym1[rt][i];   // x_r = mp[:,:OUT] - mm[:,OUT:]
            opi[off] = ym0[rt][i] + yp1[rt][i];   // x_i = mm[:,:OUT] + mp[:,OUT:]
          }
      }
    }
  }
}

extern "C" void kernel_launch(void* const* d_in, const int* in_sizes, int n_in,
                              void* d_out, int out_size, void* d_ws, size_t ws_size,
                              hipStream_t stream) {
  const float* emb = (const float*)d_in[0];
  const float* xe  = (const float*)d_in[1];
  const float* Wd0 = (const float*)d_in[2];
  const float* bd0 = (const float*)d_in[3];
  const float* W0  = (const float*)d_in[4];
  const float* b0  = (const float*)d_in[5];
  const float* Wd1 = (const float*)d_in[6];
  const float* bd1 = (const float*)d_in[7];
  const float* W1  = (const float*)d_in[8];
  const float* Wd2 = (const float*)d_in[9];
  const float* bd2 = (const float*)d_in[10];
  const float* W2  = (const float*)d_in[11];
  float* out = (float*)d_out;

  // Workspace partition (bf16 elements, 16B aligned regions)
  unsigned short* P = (unsigned short*)d_ws;
  unsigned short* pWd0 = P;                         // 128*640
  unsigned short* pWd1 = pWd0 + 128 * 640;          // 128*512
  unsigned short* pWd2 = pWd1 + 128 * 512;          // 128*384
  unsigned short* pW0  = pWd2 + 128 * 384;          // 640*640
  unsigned short* pW1  = pW0 + 640 * 640;           // 512*1024
  unsigned short* pW2  = pW1 + 512 * 1024;          // 384*768

  // Pack weights into WMMA B-fragment bf16 layout (one thread / 16 elems)
  pack_weights_kernel<<<(128 * 640 / 16 + 255) / 256, 256, 0, stream>>>(Wd0, pWd0, 128, 640);
  pack_weights_kernel<<<(128 * 512 / 16 + 255) / 256, 256, 0, stream>>>(Wd1, pWd1, 128, 512);
  pack_weights_kernel<<<(128 * 384 / 16 + 255) / 256, 256, 0, stream>>>(Wd2, pWd2, 128, 384);
  pack_weights_kernel<<<(640 * 640 / 16 + 255) / 256, 256, 0, stream>>>(W0, pW0, 640, 640);
  pack_weights_kernel<<<(512 * 1024 / 16 + 255) / 256, 256, 0, stream>>>(W1, pW1, 512, 1024);
  pack_weights_kernel<<<(384 * 768 / 16 + 255) / 256, 256, 0, stream>>>(W2, pW2, 384, 768);

  int num = in_sizes[0] / (L2S * CH);
  dim3 grid(num / TM), block(256);
  so2_conv_kernel<<<grid, block, 0, stream>>>(emb, xe, pWd0, bd0, pW0, b0,
                                              pWd1, bd1, pW1, pWd2, bd2, pW2, out);
}
